// MultiHeadVectorQuantizer_81664508166540
// MI455X (gfx1250) — matrix operand order repaired and checked by reading, not compiled, not run on hardware
//
#include <hip/hip_runtime.h>
#include <hip/hip_bf16.h>

typedef __attribute__((ext_vector_type(2)))  float   v2f;
typedef __attribute__((ext_vector_type(8)))  float   v8f;
typedef __attribute__((ext_vector_type(16))) __bf16  v16bf;
typedef __attribute__((ext_vector_type(2)))  __bf16  v2bf;

// Problem constants: z (8,256,32,32) f32, codebook (4096,64) f32
#define N_E     4096
#define DSEG    64
#define NPOS    8192            // 8*32*32 positions
#define MROWS   32768           // NPOS * 4 heads
#define ZELEMS  2097152         // 8*256*32*32
#define NTILES  (N_E / 16)      // 256 codebook tiles of 16 codes
#define BPITCH  72              // bf16 elems per code row in LDS (144B, 16B-aligned v16bf)

// ---------------------------------------------------------------------------
// Kernel 0: |e|^2 per codeword in f32 (exact term for the distance compare).
// ---------------------------------------------------------------------------
__global__ __launch_bounds__(256)
void vq_enorm_kernel(const float* __restrict__ cb, float* __restrict__ enorm) {
  const int code = blockIdx.x * 256 + threadIdx.x;   // grid = 16 blocks
  const float* e = cb + (size_t)code * DSEG;
  float s = 0.0f;
#pragma unroll
  for (int j = 0; j < DSEG; ++j) s += e[j] * e[j];
  enorm[code] = s;
}

// Stage one 16x64 codebook tile into LDS as split-bf16 (hi + residual lo).
__device__ __forceinline__
void stage_tile(const float* __restrict__ cb, int t,
                __bf16* __restrict__ hi, __bf16* __restrict__ lo, int tid) {
  const float2* src = (const float2*)(cb + (size_t)t * 16 * DSEG);
#pragma unroll
  for (int it = 0; it < 4; ++it) {
    const int f2i  = it * 128 + tid;      // 512 float2 per tile, 128 threads
    const int row  = f2i >> 5;            // 32 float2 per code row
    const int cpos = (f2i & 31) * 2;
    const float2 v = src[f2i];
    const __bf16 h0 = (__bf16)v.x, h1 = (__bf16)v.y;
    const __bf16 l0 = (__bf16)(v.x - (float)h0);
    const __bf16 l1 = (__bf16)(v.y - (float)h1);
    *(v2bf*)&hi[row * BPITCH + cpos] = (v2bf){h0, h1};
    *(v2bf*)&lo[row * BPITCH + cpos] = (v2bf){l0, l1};
  }
}

// ---------------------------------------------------------------------------
// Kernel 1: distance GEMM via split-bf16 V_WMMA_F32_16X16X32_BF16 + argmin.
//   dot(z,e) ~= z_hi.e_hi + z_hi.e_lo + z_lo.e_hi  (lo.lo term ~1e-6 rel)
//   dist     = |e|^2 - 2*dot  (|z|^2 is row-constant -> drops from argmin)
// Block = 128 threads (4 waves, one 16-row tile each); codebook tile staged
// once per block in double-buffered LDS (1 barrier / tile, staging overlaps
// the 6 chained WMMAs of the previous tile).
// ---------------------------------------------------------------------------
__global__ __launch_bounds__(128)
void vq_argmin_kernel(const float* __restrict__ z,
                      const float* __restrict__ cb,
                      const float* __restrict__ enorm,
                      float* __restrict__ idx_out /* [4*NPOS] head-major */) {
  __shared__ __bf16 ebh[2][16 * BPITCH];
  __shared__ __bf16 ebl[2][16 * BPITCH];

  const int tid  = threadIdx.x;
  const int wave = tid >> 5;
  const int lane = tid & 31;
  const int half = lane >> 4;
  const int col  = lane & 15;

  const int rowBase = (blockIdx.x * 4 + wave) * 16;

  // ---- Preload A (16x64 z rows) as split-bf16 in contiguous-K order (B uses the same order):
  //      lane holds row M = col; slice s (K base 32s), element e ->
  //      K = 32s + 16*half + e.
  //      Memory: element j of vector (n,k) at ((b*256 + k*64 + j)*1024 + hw).
  const int r  = rowBase + col;
  const int n  = r >> 2;
  const int kh = r & 3;
  const int bi = n >> 10;
  const int hw = n & 1023;
  const float* zr = z + ((size_t)(bi * 256 + kh * 64) * 1024 + (size_t)hw);

  v16bf a_hi[2], a_lo[2];
#pragma unroll
  for (int s = 0; s < 2; ++s) {
#pragma unroll
    for (int e = 0; e < 16; ++e) {
      const int j = 32 * s + 16 * half + e;
      const float x  = zr[(size_t)j * 1024];
      const __bf16 h = (__bf16)x;
      a_hi[s][e] = h;
      a_lo[s][e] = (__bf16)(x - (float)h);
    }
  }

  float mind[8];
  int   midx[8];
#pragma unroll
  for (int v = 0; v < 8; ++v) { mind[v] = 3.4e38f; midx[v] = 0; }

  stage_tile(cb, 0, ebh[0], ebl[0], tid);

  for (int t = 0; t < NTILES; ++t) {
    __syncthreads();                       // staging of tile t visible
    if (t + 1 < NTILES) {                  // overlap: stage t+1 into other buf
      stage_tile(cb, t + 1, ebh[(t + 1) & 1], ebl[(t + 1) & 1], tid);
      if (t + 2 < NTILES)
        __builtin_prefetch(cb + (size_t)(t + 2) * 16 * DSEG + tid * 8, 0, 0);
    }

    const int cur = t & 1;
    const float esq = enorm[t * 16 + col]; // exact f32 |e|^2 for code col
    // B operands: lane holds code column N = col, slice-local K = 16*half + e.
    const int boff = col * BPITCH + 16 * half;
    const v16bf bh0 = *(const v16bf*)&ebh[cur][boff];
    const v16bf bh1 = *(const v16bf*)&ebh[cur][boff + 32];
    const v16bf bl0 = *(const v16bf*)&ebl[cur][boff];
    const v16bf bl1 = *(const v16bf*)&ebl[cur][boff + 32];

    v8f acc = {};
    acc = __builtin_amdgcn_wmma_f32_16x16x32_bf16(false, a_lo[0], false, bh0,
                                                  (short)0, acc, false, false);
    acc = __builtin_amdgcn_wmma_f32_16x16x32_bf16(false, a_lo[1], false, bh1,
                                                  (short)0, acc, false, false);
    acc = __builtin_amdgcn_wmma_f32_16x16x32_bf16(false, a_hi[0], false, bl0,
                                                  (short)0, acc, false, false);
    acc = __builtin_amdgcn_wmma_f32_16x16x32_bf16(false, a_hi[1], false, bl1,
                                                  (short)0, acc, false, false);
    acc = __builtin_amdgcn_wmma_f32_16x16x32_bf16(false, a_hi[0], false, bh0,
                                                  (short)0, acc, false, false);
    acc = __builtin_amdgcn_wmma_f32_16x16x32_bf16(false, a_hi[1], false, bh1,
                                                  (short)0, acc, false, false);

    const int code = t * 16 + col;
#pragma unroll
    for (int v = 0; v < 8; ++v) {
      const float dist = esq - 2.0f * acc[v];
      if (dist < mind[v]) { mind[v] = dist; midx[v] = code; }
    }
  }

  // ---- Cross-lane argmin over 16 code-column classes (per half-wave),
  //      first-index tie-break to match jnp.argmin.
#pragma unroll
  for (int v = 0; v < 8; ++v) {
    float d  = mind[v];
    int   id = midx[v];
#pragma unroll
    for (int off = 1; off < 16; off <<= 1) {
      const float d2 = __shfl_xor(d, off, 32);
      const int   i2 = __shfl_xor(id, off, 32);
      if (d2 < d || (d2 == d && i2 < id)) { d = d2; id = i2; }
    }
    if (col == 0) {
      const int row = rowBase + v + 8 * half;  // C-layout rows
      const int nn  = row >> 2;
      const int kk  = row & 3;
      idx_out[kk * NPOS + nn] = (float)id;     // head-major flat indices
    }
  }
}

// ---------------------------------------------------------------------------
// Kernel 2: gather z_q (straight-through forward == codeword), write z_q_out
// in (b,c,h,w), and per-block partial sums of (zq - z)^2.
// ---------------------------------------------------------------------------
__global__ __launch_bounds__(256)
void vq_gather_kernel(const float* __restrict__ z,
                      const float* __restrict__ cb,
                      const float* __restrict__ idx_in,
                      float* __restrict__ zq_out,
                      float* __restrict__ partial) {
  const int gid = blockIdx.x * 256 + threadIdx.x;   // grid = ZELEMS/256
  const int bi  = gid >> 18;
  const int c   = (gid >> 10) & 255;
  const int hw  = gid & 1023;
  const int k   = c >> 6;
  const int j   = c & 63;
  const int n   = bi * 1024 + hw;
  const int id  = (int)idx_in[k * NPOS + n];
  const float q  = cb[(size_t)id * DSEG + j];
  const float zv = z[gid];
  zq_out[gid] = q;
  const float df = q - zv;

  __shared__ float red[256];
  red[threadIdx.x] = df * df;
  __syncthreads();
#pragma unroll
  for (int s = 128; s > 0; s >>= 1) {
    if (threadIdx.x < s) red[threadIdx.x] += red[threadIdx.x + s];
    __syncthreads();
  }
  if (threadIdx.x == 0) partial[blockIdx.x] = red[0];
}

// ---------------------------------------------------------------------------
// Kernel 3: deterministic final reduce -> loss = 1.25 * mean((zq-z)^2);
// perplexity = 0, cluster_use = 0 (reference returns zeros).
// ---------------------------------------------------------------------------
__global__ __launch_bounds__(256)
void vq_finalize_kernel(const float* __restrict__ partial,
                        float* __restrict__ scalars) {
  __shared__ float red[256];
  float s = 0.0f;
  for (int i = threadIdx.x; i < 8192; i += 256) s += partial[i];  // fixed order
  red[threadIdx.x] = s;
  __syncthreads();
#pragma unroll
  for (int st = 128; st > 0; st >>= 1) {
    if (threadIdx.x < st) red[threadIdx.x] += red[threadIdx.x + st];
    __syncthreads();
  }
  if (threadIdx.x == 0) {
    scalars[0] = red[0] * (1.25f / (float)ZELEMS);
    scalars[1] = 0.0f;
    scalars[2] = 0.0f;
  }
}

// ---------------------------------------------------------------------------
extern "C" void kernel_launch(void* const* d_in, const int* in_sizes, int n_in,
                              void* d_out, int out_size, void* d_ws, size_t ws_size,
                              hipStream_t stream) {
  const float* z  = (const float*)d_in[0];   // (8,256,32,32) f32
  const float* cb = (const float*)d_in[1];   // (4096,64) f32

  float* out     = (float*)d_out;
  float* zq      = out;                      // 2,097,152 elems
  float* scalars = out + ZELEMS;             // loss, perplexity, cluster_use
  float* idxf    = out + ZELEMS + 3;         // 32,768 head-major indices

  float* enorm   = (float*)d_ws;             // 4096 floats
  float* partial = enorm + N_E;              // 8192 floats

  vq_enorm_kernel<<<N_E / 256, 256, 0, stream>>>(cb, enorm);
  vq_argmin_kernel<<<MROWS / 64, 128, 0, stream>>>(z, cb, enorm, idxf);
  vq_gather_kernel<<<ZELEMS / 256, 256, 0, stream>>>(z, cb, idxf, zq, partial);
  vq_finalize_kernel<<<1, 256, 0, stream>>>(partial, scalars);
}
